// GraphNet_45140106281309
// MI455X (gfx1250) — matrix operand order repaired and checked by reading, not compiled, not run on hardware
//
#include <hip/hip_runtime.h>
#include <hip/hip_bf16.h>
#include <math.h>

typedef _Float16 v16h __attribute__((ext_vector_type(16)));
typedef _Float16 v8h  __attribute__((ext_vector_type(8)));
typedef float    v8f  __attribute__((ext_vector_type(8)));

#define NNODES  100000
#define NEDGES  3200000
#define NGRAPHS 512
#define FSTRIDE 32   // padded f32 feature stride: 12.8MB buffers, L2-resident

// ---------------- activations (fused into the f16 A-pack pass) --------------
__device__ __forceinline__ float gelu_f(float x) {
    const float c = 0.7978845608028654f; // sqrt(2/pi), jax.nn.gelu approximate=True
    float x3 = x * x * x;
    return 0.5f * x * (1.0f + tanhf(c * (x + 0.044715f * x3)));
}
__device__ __forceinline__ float act_f(float x, int mode) {
    if (mode == 1) return gelu_f(x);
    if (mode == 2) return fmaxf(x, 0.0f);
    return x;
}

// ---------------- GCN symmetric norm precompute ------------------------------
__global__ void k_init_deg(float* deg, int n) {
    int i = blockIdx.x * blockDim.x + threadIdx.x;
    if (i < n) deg[i] = 1.0f;                      // self-loop contributes 1
}
__global__ void k_deg_accum(const int* __restrict__ col, float* __restrict__ deg, int e) {
    int i = blockIdx.x * blockDim.x + threadIdx.x;
    if (i < e) atomicAdd(&deg[col[i]], 1.0f);
}
__global__ void k_dinv(float* deg, int n) {
    int i = blockIdx.x * blockDim.x + threadIdx.x;
    if (i < n) { float d = deg[i]; deg[i] = (d > 0.0f) ? rsqrtf(d) : 0.0f; }
}
__global__ void k_norm(const int* __restrict__ row, const int* __restrict__ col,
                       const float* __restrict__ dinv, float* __restrict__ norm, int e) {
    int i = blockIdx.x * blockDim.x + threadIdx.x;
    if (i < e) norm[i] = dinv[row[i]] * dinv[col[i]];
}

// ---------------- operand packing -------------------------------------------
// A-pack: f32 (stride xs) -> f16 padded row-major [node][KPAD], activation fused,
// zero fill for k >= fin. Makes GEMM A-loads unconditional 16B vector loads.
__global__ void k_packA(const float* __restrict__ X, int xs, int fin, int act_mode,
                        _Float16* __restrict__ XP, int kpad_shift, int total) {
    int idx = blockIdx.x * blockDim.x + threadIdx.x;
    if (idx >= total) return;
    int node = idx >> kpad_shift;
    int k    = idx & ((1 << kpad_shift) - 1);
    float v = (k < fin) ? X[(size_t)node * xs + k] : 0.0f;
    XP[idx] = (_Float16)act_f(v, act_mode);
}

// B-pack: W (fin x fout, f32) -> per-lane fragment order [s][t][lane][16 halves].
// Fragment layout for 16-bit B 32x16 (ISA 7.12.2): lane n = L&15;
// half h -> k = k0 + 8*(L>=16) + (h<8 ? h : h+8).
__global__ void k_packB(const float* __restrict__ W, int fin, int fout,
                        int nks, int nct, _Float16* __restrict__ BP) {
    int tid = blockIdx.x * blockDim.x + threadIdx.x;     // (s*nct + t)*32 + lane
    int tot = nks * nct * 32;
    if (tid >= tot) return;
    int lane = tid & 31;
    int st   = tid >> 5;
    int t    = st % nct;
    int s    = st / nct;
    int ncol = t * 16 + (lane & 15);
    int kb   = s * 32 + (lane >> 4) * 8;
    #pragma unroll
    for (int h = 0; h < 16; ++h) {
        int k = kb + (h < 8 ? h : h + 8);
        float v = (k < fin && ncol < fout) ? W[k * fout + ncol] : 0.0f;
        BP[(size_t)tid * 16 + h] = (_Float16)v;
    }
}

// ---------------- dense linear via V_WMMA_F32_16X16X32_F16 -------------------
// One wave computes a 16-node x (NCT*16)-feature tile. All bounds compile-time:
// fully unrolled, branch-free inner loop, unconditional vector loads/stores.
// A-frag (16x32 f16): lane m=L&15; halves e<8 -> k=k0+8*hi+e ; e>=8 -> k=k0+16+8*hi+(e-8)
// C/D (16x16 f32): lane n=L&15; VGPR r -> m = r + 8*hi
template<int KPAD, int NKS, int NCT>
__global__ void k_gemm_wmma(const _Float16* __restrict__ XP,
                            const _Float16* __restrict__ BP,
                            float* __restrict__ H, int nTiles) {
    const int lane = threadIdx.x & 31;
    const int wave = threadIdx.x >> 5;
    const int wpb  = blockDim.x >> 5;
    const int tile = blockIdx.x * wpb + wave;
    if (tile >= nTiles) return;                 // wave-uniform

    const int m  = lane & 15;
    const int hi = lane >> 4;
    const int grp = hi * 8;

    v8f c[NCT];
    #pragma unroll
    for (int t = 0; t < NCT; ++t)
        #pragma unroll
        for (int r = 0; r < 8; ++r) c[t][r] = 0.0f;

    const _Float16* xrow = XP + (size_t)(tile * 16 + m) * KPAD;

    #pragma unroll
    for (int s = 0; s < NKS; ++s) {
        // A fragment: two unconditional 16B loads (8 halves each)
        v8h alo = *(const v8h*)(xrow + s * 32 + grp);
        v8h ahi = *(const v8h*)(xrow + s * 32 + 16 + grp);
        v16h a = __builtin_shufflevector(alo, ahi, 0, 1, 2, 3, 4, 5, 6, 7,
                                                    8, 9, 10, 11, 12, 13, 14, 15);
        #pragma unroll
        for (int t = 0; t < NCT; ++t) {
            // B fragment: one unconditional 32B load in pre-packed lane order
            v16h b = *(const v16h*)(BP + ((size_t)((s * NCT + t) * 32 + lane)) * 16);
            c[t] = __builtin_amdgcn_wmma_f32_16x16x32_f16(
                false, a, false, b, (short)0, c[t], false, false);
        }
    }
    // Unconditional stores: padded columns are never read downstream.
    #pragma unroll
    for (int t = 0; t < NCT; ++t)
        #pragma unroll
        for (int r = 0; r < 8; ++r)
            H[(size_t)(tile * 16 + r + hi * 8) * FSTRIDE + t * 16 + (lane & 15)] = c[t][r];
}

// ---------------- aggregation: out = b + dinv^2*H + scatter(norm*H[col]) -----
__global__ void k_agg_init(const float* __restrict__ H, const float* __restrict__ dinv,
                           const float* __restrict__ bias, float* __restrict__ out,
                           int fout, int n) {
    int idx = blockIdx.x * blockDim.x + threadIdx.x;
    int i = idx >> 5, f = idx & 31;
    if (i < n && f < fout) {
        float di = dinv[i];
        out[(size_t)i * FSTRIDE + f] = bias[f] + di * di * H[(size_t)i * FSTRIDE + f];
    }
}

template<int F>
__global__ void k_agg_edges(const float* __restrict__ H, const int* __restrict__ row,
                            const int* __restrict__ col, const float* __restrict__ norm,
                            float* __restrict__ out, int e) {
    int i = blockIdx.x * blockDim.x + threadIdx.x;
    if (i >= e) return;
    const int   r = row[i], cc = col[i];
    const float w = norm[i];
    const float* __restrict__ hp = H   + (size_t)cc * FSTRIDE;  // L2-resident gather
    float*       __restrict__ op = out + (size_t)r  * FSTRIDE;  // L2-resident atomics
    float v[F];
    #pragma unroll
    for (int f = 0; f < F; ++f) v[f] = w * hp[f];
    #pragma unroll
    for (int f = 0; f < F; ++f) atomicAdd(&op[f], v[f]);
}

// ---------------- graph mean pool -------------------------------------------
__global__ void k_pool_zero(float* s, float* cnt) {
    int g = blockIdx.x * blockDim.x + threadIdx.x;
    if (g < NGRAPHS) { s[g] = 0.0f; cnt[g] = 0.0f; }
}
__global__ void k_pool_accum(const float* __restrict__ A4, const int* __restrict__ batch,
                             float* __restrict__ s, float* __restrict__ cnt, int n) {
    int i = blockIdx.x * blockDim.x + threadIdx.x;
    if (i < n) {
        int g = batch[i];
        atomicAdd(&s[g], A4[(size_t)i * FSTRIDE]);
        atomicAdd(&cnt[g], 1.0f);
    }
}
__global__ void k_pool_final(const float* s, const float* cnt, float* out) {
    int g = blockIdx.x * blockDim.x + threadIdx.x;
    if (g < NGRAPHS) out[g] = s[g] / fmaxf(cnt[g], 1.0f);
}

// ---------------- launch -----------------------------------------------------
extern "C" void kernel_launch(void* const* d_in, const int* in_sizes, int n_in,
                              void* d_out, int out_size, void* d_ws, size_t ws_size,
                              hipStream_t stream) {
    const float* x     = (const float*)d_in[0];
    const int*   ei    = (const int*)d_in[1];
    const int*   batch = (const int*)d_in[2];
    const float* W1 = (const float*)d_in[3];  const float* b1 = (const float*)d_in[4];
    const float* W2 = (const float*)d_in[5];  const float* b2 = (const float*)d_in[6];
    const float* W3 = (const float*)d_in[7];  const float* b3 = (const float*)d_in[8];
    const float* W4 = (const float*)d_in[9];  const float* b4 = (const float*)d_in[10];
    float* out = (float*)d_out;

    const int* row = ei;            // targets
    const int* col = ei + NEDGES;   // sources

    // ws layout: norm[E] f32 | dinv[N] f32 | bufH[N*32] f32 | bufA[N*32] f32 |
    //            psum/pcnt | XP[N*64] f16 | BP[4K] f16
    float* ws   = (float*)d_ws;
    float* norm = ws;
    float* dinv = norm + NEDGES;
    float* bufH = dinv + NNODES;
    float* bufA = bufH + (size_t)NNODES * FSTRIDE;
    float* psum = bufA + (size_t)NNODES * FSTRIDE;
    float* pcnt = psum + NGRAPHS;
    _Float16* XP = (_Float16*)(pcnt + NGRAPHS);             // N*64 halves (8B-aligned ok, rows 64/128B)
    _Float16* BP = XP + (size_t)NNODES * 64;                // <= 2*2*32*16 halves

    const int B = 256;
    const int gN  = (NNODES + B - 1) / B;
    const int gE  = (NEDGES + B - 1) / B;
    const int gNF = (NNODES * FSTRIDE + B - 1) / B;
    const int gP64 = (NNODES * 64 + B - 1) / B;
    const int gP32 = (NNODES * 32 + B - 1) / B;

    // normalization (recomputed every call -> deterministic, no cross-call state)
    k_init_deg <<<gN, B, 0, stream>>>(dinv, NNODES);
    k_deg_accum<<<gE, B, 0, stream>>>(col, dinv, NEDGES);
    k_dinv     <<<gN, B, 0, stream>>>(dinv, NNODES);
    k_norm     <<<gE, B, 0, stream>>>(row, col, dinv, norm, NEDGES);

    const int TILES = (NNODES + 15) / 16;      // 6250 (N divisible by 16)
    const int WPB   = 8;                       // 8 wave32s per block
    dim3 gg((TILES + WPB - 1) / WPB), gb(WPB * 32);

    // layer 1: H1 = x @ W1 (Fin=37 -> KPAD=64, NKS=2; Fout=25 -> NCT=2)
    k_packA<<<gP64, B, 0, stream>>>(x, 37, 37, 0, XP, 6, NNODES * 64);
    k_packB<<<1, B, 0, stream>>>(W1, 37, 25, 2, 2, BP);
    k_gemm_wmma<64, 2, 2><<<gg, gb, 0, stream>>>(XP, BP, bufH, TILES);
    k_agg_init <<<gNF, B, 0, stream>>>(bufH, dinv, b1, bufA, 25, NNODES);
    k_agg_edges<25><<<gE, B, 0, stream>>>(bufH, row, col, norm, bufA, NEDGES);

    // layer 2: gelu fused into A-pack (Fin=25 -> KPAD=32; Fout=18 -> NCT=2)
    k_packA<<<gP32, B, 0, stream>>>(bufA, FSTRIDE, 25, 1, XP, 5, NNODES * 32);
    k_packB<<<1, B, 0, stream>>>(W2, 25, 18, 1, 2, BP);
    k_gemm_wmma<32, 1, 2><<<gg, gb, 0, stream>>>(XP, BP, bufH, TILES);
    k_agg_init <<<gNF, B, 0, stream>>>(bufH, dinv, b2, bufA, 18, NNODES);
    k_agg_edges<18><<<gE, B, 0, stream>>>(bufH, row, col, norm, bufA, NEDGES);

    // layer 3: relu fused (Fin=18; Fout=12 -> NCT=1)
    k_packA<<<gP32, B, 0, stream>>>(bufA, FSTRIDE, 18, 2, XP, 5, NNODES * 32);
    k_packB<<<1, B, 0, stream>>>(W3, 18, 12, 1, 1, BP);
    k_gemm_wmma<32, 1, 1><<<gg, gb, 0, stream>>>(XP, BP, bufH, TILES);
    k_agg_init <<<gNF, B, 0, stream>>>(bufH, dinv, b3, bufA, 12, NNODES);
    k_agg_edges<12><<<gE, B, 0, stream>>>(bufH, row, col, norm, bufA, NEDGES);

    // layer 4: gelu fused (Fin=12; Fout=1 -> NCT=1)
    k_packA<<<gP32, B, 0, stream>>>(bufA, FSTRIDE, 12, 1, XP, 5, NNODES * 32);
    k_packB<<<1, B, 0, stream>>>(W4, 12, 1, 1, 1, BP);
    k_gemm_wmma<32, 1, 1><<<gg, gb, 0, stream>>>(XP, BP, bufH, TILES);
    k_agg_init <<<gNF, B, 0, stream>>>(bufH, dinv, b4, bufA, 1, NNODES);
    k_agg_edges<1><<<gE, B, 0, stream>>>(bufH, row, col, norm, bufA, NEDGES);

    // scatter-mean over graphs
    k_pool_zero <<<2, B, 0, stream>>>(psum, pcnt);
    k_pool_accum<<<gN, B, 0, stream>>>(bufA, batch, psum, pcnt, NNODES);
    k_pool_final<<<2, B, 0, stream>>>(psum, pcnt, out);
}
